// ExomaAttentionDecoder_21406117003537
// MI455X (gfx1250) — matrix operand order repaired and checked by reading, not compiled, not run on hardware
//
#include <hip/hip_runtime.h>

// =====================================================================
// MI455X (gfx1250, CDNA5, wave32) decoder layer:
//   rmsnorm -> QKV GEMM -> DCT linear attention (4x batched 256^3 GEMMs
//   + 2 softmaxes) -> O-proj + residual -> rmsnorm -> FFF tree.
// All GEMMs run on v_wmma_f32_16x16x32_f16 (f16 operands, f32 accum).
// Compute-bound: ~1.17 TFLOP vs ~1GB HBM traffic (~45us @ 23.3 TB/s).
// GEMM tile fill uses the CDNA5 async global->LDS path (ASYNCcnt):
// probe showed __builtin_amdgcn_global_load_async_to_lds_b128 takes
// (int4 AS1*, int4 AS3*, imm offset, imm cpol).
// =====================================================================

typedef __attribute__((ext_vector_type(16))) _Float16 v16h;
typedef __attribute__((ext_vector_type(8)))  _Float16 v8h;
typedef __attribute__((ext_vector_type(8)))  float    v8f;
typedef __attribute__((ext_vector_type(4)))  int      v4i;

#define DIMX   4096
#define HDX    256
#define NHEAD  16
#define NBATCH 32
#define SEQ    256
#define ROWS   (NBATCH * SEQ)       // 8192
#define QKV_N  (3 * NHEAD * HDX)    // 12288
#define FDEPTH 12

// ---------------- workspace layout (bytes) -------------------------
// peak ~608 MB, regions reused once their producer/consumer is done.
static const size_t OFF_W16    = 0;                 // qkv_w f16 (100663296B), later o_w f16
static const size_t OFF_HA     = 100663296;         // rms1 f16 (67108864B), later scores f16
static const size_t OFF_Q      = 167772160;         // q   f16 [b,h,t,d]  (67108864B)
static const size_t OFF_KT     = 234881024;         // k^T f16 [b,h,d,s]  (67108864B)
static const size_t OFF_VT     = 301989888;         // v^T f16 [b,h,d,s]  (67108864B)
static const size_t OFF_PROJT  = 369098752;         // proj^T f16 (131072B)
static const size_t OFF_F32SCR = 369229824;         // logits / m_f32 (134217728B)
static const size_t OFF_QUERYT = 503447552;         // query^T f16 (67108864B)
static const size_t OFF_KEYM   = 570556416;         // keym f16 (67108864B), later attnA

constexpr int MODE_F32   = 0;
constexpr int MODE_F16   = 1;
constexpr int MODE_RESID = 2;
constexpr int MODE_QKV   = 3;
constexpr int MODE_ATTN  = 4;

union V16U { v16h v; v8h h[2]; };

// -------- CDNA5 async global->LDS (GLOBAL_LOAD_ASYNC_TO_LDS_B128) ----
#if defined(__gfx1250__) && __has_builtin(__builtin_amdgcn_global_load_async_to_lds_b128)
#define USE_ASYNC_LDS 1
__device__ __forceinline__ void async_copy_b128(const _Float16* g, _Float16* l) {
  // low 32 bits of a generic LDS pointer == workgroup-relative LDS offset
  __builtin_amdgcn_global_load_async_to_lds_b128(
      (v4i __attribute__((address_space(1)))*)(uintptr_t)g,
      (v4i __attribute__((address_space(3)))*)(unsigned)(uintptr_t)l,
      /*offset=*/0, /*cpol=*/0);
}
__device__ __forceinline__ void wait_async_le4(void) {
#if __has_builtin(__builtin_amdgcn_s_wait_asynccnt)
  __builtin_amdgcn_s_wait_asynccnt(4);
#else
  asm volatile("s_wait_asynccnt 0x4" ::: "memory");
#endif
}
__device__ __forceinline__ void wait_async_le0(void) {
#if __has_builtin(__builtin_amdgcn_s_wait_asynccnt)
  __builtin_amdgcn_s_wait_asynccnt(0);
#else
  asm volatile("s_wait_asynccnt 0x0" ::: "memory");
#endif
}
#else
#define USE_ASYNC_LDS 0
#endif

// =====================================================================
// Generic NT GEMM on WMMA: C[M,N] = A[M,K] * B[N,K]^T  (f16 in, f32 acc)
// 128x128x32 tiles, 8 waves (4x2), 2x4 accum tiles per wave,
// double-buffered LDS; tile fill via async-to-LDS when available.
// =====================================================================
template <int MODE>
__global__ __launch_bounds__(256) void gemm_nt_wmma(
    const _Float16* __restrict__ A, const _Float16* __restrict__ B,
    int M, int N, int K,
    long aBatch, long bBatch, long cBatch,
    float* __restrict__ cOutF32, _Float16* __restrict__ cOutF16,
    const float* __restrict__ resid, const int* __restrict__ kvidx,
    _Float16* __restrict__ qbuf, _Float16* __restrict__ kbuf,
    _Float16* __restrict__ vbuf)
{
  constexpr int BM = 128, BN = 128, BK = 32;
  __shared__ _Float16 sA[2][BM][BK];   // 8KB x2
  __shared__ _Float16 sB[2][BN][BK];   // 8KB x2

  const int tid  = threadIdx.x;
  const int lane = tid & 31;
  const int wave = tid >> 5;          // 8 waves
  const int wm   = wave >> 1;         // 0..3 -> 32-row slice
  const int wn   = wave & 1;          // 0..1 -> 64-col slice
  const int bm0  = blockIdx.y * BM;
  const int bn0  = blockIdx.x * BN;
  const int z    = blockIdx.z;

  const _Float16* Ab = A + (long)z * aBatch;
  const _Float16* Bb = B + (long)z * bBatch;

  // staging: 16 halves (2x 16B) per thread per matrix per stage
  const int ldRow = tid >> 1;
  const int ldCol = (tid & 1) * 16;
  const _Float16* gA = Ab + (long)(bm0 + ldRow) * K + ldCol;
  const _Float16* gB = Bb + (long)(bn0 + ldRow) * K + ldCol;

  const int nk = K / BK;

#if USE_ASYNC_LDS
  // prologue: stage 0 straight into LDS via the async-DMA path
  async_copy_b128(gA,     &sA[0][ldRow][ldCol]);
  async_copy_b128(gA + 8, &sA[0][ldRow][ldCol + 8]);
  async_copy_b128(gB,     &sB[0][ldRow][ldCol]);
  async_copy_b128(gB + 8, &sB[0][ldRow][ldCol + 8]);
#else
  v8h ra0 = *(const v8h*)(gA);
  v8h ra1 = *(const v8h*)(gA + 8);
  v8h rb0 = *(const v8h*)(gB);
  v8h rb1 = *(const v8h*)(gB + 8);
  *(v8h*)&sA[0][ldRow][ldCol]     = ra0;
  *(v8h*)&sA[0][ldRow][ldCol + 8] = ra1;
  *(v8h*)&sB[0][ldRow][ldCol]     = rb0;
  *(v8h*)&sB[0][ldRow][ldCol + 8] = rb1;
  __syncthreads();
#endif

  v8f acc[2][4];
  {
    v8f z8 = {};
#pragma unroll
    for (int i = 0; i < 2; ++i)
#pragma unroll
      for (int j = 0; j < 4; ++j) acc[i][j] = z8;
  }

  for (int kb = 0; kb < nk; ++kb) {
    const int buf = kb & 1;
    const bool more = (kb + 1 < nk);

#if USE_ASYNC_LDS
    if (more) {  // fill next buffer, overlapped with this step's WMMAs
      const _Float16* nA = gA + (kb + 1) * BK;
      const _Float16* nB = gB + (kb + 1) * BK;
      async_copy_b128(nA,     &sA[buf ^ 1][ldRow][ldCol]);
      async_copy_b128(nA + 8, &sA[buf ^ 1][ldRow][ldCol + 8]);
      async_copy_b128(nB,     &sB[buf ^ 1][ldRow][ldCol]);
      async_copy_b128(nB + 8, &sB[buf ^ 1][ldRow][ldCol + 8]);
      wait_async_le4();   // stage kb's 4 ops landed (in-order completion)
    } else {
      wait_async_le0();
    }
    __syncthreads();      // cross-wave visibility of stage kb
#else
    if (more) {
      const _Float16* nA = gA + (kb + 1) * BK;
      const _Float16* nB = gB + (kb + 1) * BK;
      if (kb + 2 < nk) {  // gfx1250 global_prefetch_b8
        __builtin_prefetch(gA + (kb + 2) * BK, 0, 1);
        __builtin_prefetch(gB + (kb + 2) * BK, 0, 1);
      }
      ra0 = *(const v8h*)(nA);
      ra1 = *(const v8h*)(nA + 8);
      rb0 = *(const v8h*)(nB);
      rb1 = *(const v8h*)(nB + 8);
    }
#endif

    // A fragment (16x32, MxK): lane<16 -> K {0..7,16..23}, lane>=16 -> +8
    V16U af[2], bf[4];
    const int arow = wm * 32 + (lane & 15);
    const int kh8  = (lane >> 4) * 8;
#pragma unroll
    for (int i = 0; i < 2; ++i) {
      af[i].h[0] = *(const v8h*)&sA[buf][arow + i * 16][kh8];
      af[i].h[1] = *(const v8h*)&sA[buf][arow + i * 16][16 + kh8];
    }
    // B fragment (32x16, KxN) from NT storage: lane&15 = n, lane>>4 picks K half
    const int brow = wn * 64 + (lane & 15);
  const int kg   = (lane >> 4) * 8;
#pragma unroll
    for (int j = 0; j < 4; ++j) {
      bf[j].h[0] = *(const v8h*)&sB[buf][brow + j * 16][kg];
      bf[j].h[1] = *(const v8h*)&sB[buf][brow + j * 16][16 + kg];
    }

#pragma unroll
    for (int i = 0; i < 2; ++i)
#pragma unroll
      for (int j = 0; j < 4; ++j)
        acc[i][j] = __builtin_amdgcn_wmma_f32_16x16x32_f16(
            false, af[i].v, false, bf[j].v, (short)0, acc[i][j], false, false);

#if USE_ASYNC_LDS
    if (more) __syncthreads();  // reads of buf done before next stage overwrites it
#else
    if (more) {
      __syncthreads();
      *(v8h*)&sA[buf ^ 1][ldRow][ldCol]     = ra0;
      *(v8h*)&sA[buf ^ 1][ldRow][ldCol + 8] = ra1;
      *(v8h*)&sB[buf ^ 1][ldRow][ldCol]     = rb0;
      *(v8h*)&sB[buf ^ 1][ldRow][ldCol + 8] = rb1;
      __syncthreads();
    }
#endif
  }

  // ---- epilogue: C/D layout VGPR r -> M=r (lanes 0-15) / r+8 (16-31), N=lane&15
  const int rowOff = (lane >> 4) * 8;
  const int colOff = lane & 15;
#pragma unroll
  for (int i = 0; i < 2; ++i) {
#pragma unroll
    for (int j = 0; j < 4; ++j) {
#pragma unroll
      for (int r = 0; r < 8; ++r) {
        const long gr = bm0 + wm * 32 + i * 16 + rowOff + r;
        const long gc = bn0 + wn * 64 + j * 16 + colOff;
        const float val = acc[i][j][r];
        if constexpr (MODE == MODE_F32) {
          cOutF32[(long)z * cBatch + gr * N + gc] = val;
        } else if constexpr (MODE == MODE_F16) {
          cOutF16[(long)z * cBatch + gr * N + gc] = (_Float16)val;
        } else if constexpr (MODE == MODE_RESID) {
          const long off = gr * (long)N + gc;
          cOutF32[off] = resid[off] + val;
        } else if constexpr (MODE == MODE_QKV) {
          // gr = b*256 + t ; gc in [0,12288): sect | head | d
          const int b  = (int)(gr >> 8), t = (int)(gr & 255);
          const int sect = (int)(gc >> 12);
          const int hh = (int)((gc >> 8) & 15), d = (int)(gc & 255);
          const long hb = ((long)b * NHEAD + hh) << 16;  // *65536
          if (sect == 0) {
            qbuf[hb + (long)t * HDX + d] = (_Float16)val;  // q[b,h,t,d]
          } else {
            const int slot = kvidx[t];                      // cache write index
            _Float16* dst = (sect == 1) ? kbuf : vbuf;      // kT/vT [b,h,d,slot]
            dst[hb + (long)d * HDX + slot] = (_Float16)val;
          }
        } else {  // MODE_ATTN: z = b*16+h, write attn[b, i(token)=gr, h, j=gc]
          const int b = z >> 4, hh = z & 15;
          cOutF16[(((long)b * SEQ + gr) * NHEAD + hh) * HDX + gc] = (_Float16)val;
        }
      }
    }
  }
}

// =====================================================================
// RMSNorm: one workgroup per row of 4096; out f16 (pre-GEMM) or f32.
// =====================================================================
template <typename T>
__global__ __launch_bounds__(256) void rmsnorm_kernel(
    const float* __restrict__ x, const float* __restrict__ w, T* __restrict__ out)
{
  __shared__ float red[256];
  const int tid = threadIdx.x;
  const long base = (long)blockIdx.x * DIMX;
  float v[16];
  float ss = 0.f;
#pragma unroll
  for (int j = 0; j < 16; ++j) {
    v[j] = x[base + j * 256 + tid];
    ss += v[j] * v[j];
  }
  red[tid] = ss;
  __syncthreads();
  for (int s = 128; s > 0; s >>= 1) {
    if (tid < s) red[tid] += red[tid + s];
    __syncthreads();
  }
  const float scale = rsqrtf(red[0] / (float)DIMX + 1e-6f);
#pragma unroll
  for (int j = 0; j < 16; ++j)
    out[base + j * 256 + tid] = (T)(v[j] * scale * (1.f + w[j * 256 + tid]));
}

// =====================================================================
// Row softmax over 256 elems; optional transposed f16 store (query^T).
// =====================================================================
__global__ __launch_bounds__(256) void softmax256_kernel(
    const float* __restrict__ in, _Float16* __restrict__ out, int transposeOut)
{
  __shared__ float red[256];
  const int tid = threadIdx.x;
  const long row = blockIdx.x;
  const float v = in[(row << 8) + tid];
  red[tid] = v;
  __syncthreads();
  for (int s = 128; s > 0; s >>= 1) {
    if (tid < s) red[tid] = fmaxf(red[tid], red[tid + s]);
    __syncthreads();
  }
  const float mx = red[0];
  __syncthreads();
  const float e = __expf(v - mx);
  red[tid] = e;
  __syncthreads();
  for (int s = 128; s > 0; s >>= 1) {
    if (tid < s) red[tid] += red[tid + s];
    __syncthreads();
  }
  const float o = e / red[0];
  const long batch = row >> 8;
  const long r = row & 255;
  if (transposeOut)
    out[(batch << 16) + (long)tid * 256 + r] = (_Float16)o;
  else
    out[(row << 8) + tid] = (_Float16)o;
}

// =====================================================================
// Elementwise f32 -> f16 (weights), and proj -> proj^T f16.
// =====================================================================
__global__ __launch_bounds__(256) void f32_to_f16_kernel(
    const float* __restrict__ in, _Float16* __restrict__ out, long n)
{
  for (long i = blockIdx.x * 256L + threadIdx.x; i < n; i += (long)gridDim.x * 256L)
    out[i] = (_Float16)in[i];
}

__global__ __launch_bounds__(256) void proj_transpose_kernel(
    const float* __restrict__ in, _Float16* __restrict__ out)
{
  out[(long)blockIdx.x * 256 + threadIdx.x] =
      (_Float16)in[(long)threadIdx.x * 256 + blockIdx.x];
}

// =====================================================================
// FastFeedForward tree: block per row; x,y in registers, lambda via
// block reduction (node path is block-uniform). out += y.
// =====================================================================
__global__ __launch_bounds__(256) void fff_kernel(
    const float* __restrict__ m, const float* __restrict__ Xw,
    const float* __restrict__ Yw, float* __restrict__ out)
{
  __shared__ float red[256];
  const int tid = threadIdx.x;
  const long base = (long)blockIdx.x * DIMX;
  float x[16], y[16];
#pragma unroll
  for (int j = 0; j < 16; ++j) {
    x[j] = m[base + j * 256 + tid];
    y[j] = 0.f;
  }
  int node = 0;
  for (int d = 0; d < FDEPTH; ++d) {
    const long nb = (long)node * DIMX;
    float p = 0.f;
#pragma unroll
    for (int j = 0; j < 16; ++j) p += x[j] * Xw[nb + j * 256 + tid];
    red[tid] = p;
    __syncthreads();
    for (int s = 128; s > 0; s >>= 1) {
      if (tid < s) red[tid] += red[tid + s];
      __syncthreads();
    }
    const float lam = red[0];
    __syncthreads();
#pragma unroll
    for (int j = 0; j < 16; ++j) y[j] += lam * Yw[nb + j * 256 + tid];
    node = node * 2 + 1 + (lam > 0.f ? 1 : 0);
  }
#pragma unroll
  for (int j = 0; j < 16; ++j) out[base + j * 256 + tid] += y[j];
}

// =====================================================================
extern "C" void kernel_launch(void* const* d_in, const int* in_sizes, int n_in,
                              void* d_out, int out_size, void* d_ws, size_t ws_size,
                              hipStream_t stream)
{
  (void)in_sizes; (void)n_in; (void)out_size; (void)ws_size;
  const float* hidden = (const float*)d_in[0];
  const int*   kvidx  = (const int*)d_in[1];
  // d_in[2]=k_cache (zeros), d_in[3]=v_cache (zeros), d_in[4]=mask (unused)
  const float* qkv_w  = (const float*)d_in[5];
  const float* o_w    = (const float*)d_in[6];
  const float* proj   = (const float*)d_in[7];
  const float* ln1    = (const float*)d_in[8];
  const float* ln2    = (const float*)d_in[9];
  const float* Xw     = (const float*)d_in[10];
  const float* Yw     = (const float*)d_in[11];
  float* out = (float*)d_out;
  char*  ws  = (char*)d_ws;

  _Float16* W16      = (_Float16*)(ws + OFF_W16);
  _Float16* hA       = (_Float16*)(ws + OFF_HA);
  _Float16* qbuf     = (_Float16*)(ws + OFF_Q);
  _Float16* kbuf     = (_Float16*)(ws + OFF_KT);
  _Float16* vbuf     = (_Float16*)(ws + OFF_VT);
  _Float16* projT    = (_Float16*)(ws + OFF_PROJT);
  float*    f32scr   = (float*)(ws + OFF_F32SCR);
  _Float16* queryT   = (_Float16*)(ws + OFF_QUERYT);
  _Float16* keym     = (_Float16*)(ws + OFF_KEYM);
  _Float16* scores16 = hA;    // region reuse (rms1 output dead)
  _Float16* attnA    = keym;  // region reuse (keym dead after scores GEMM)

  // cache semantics: unwritten kv slots must be zero
  (void)hipMemsetAsync(kbuf, 0, (size_t)67108864, stream);
  (void)hipMemsetAsync(vbuf, 0, (size_t)67108864, stream);

  // 1) weights -> f16, rmsnorm1 -> f16 activations
  f32_to_f16_kernel<<<4096, 256, 0, stream>>>(qkv_w, W16, (long)QKV_N * DIMX);
  rmsnorm_kernel<_Float16><<<ROWS, 256, 0, stream>>>(hidden, ln1, hA);

  // 2) QKV GEMM (8192 x 12288 x 4096) with split/scatter epilogue
  gemm_nt_wmma<MODE_QKV><<<dim3(QKV_N / 128, ROWS / 128, 1), 256, 0, stream>>>(
      hA, W16, ROWS, QKV_N, DIMX, 0, 0, 0,
      nullptr, nullptr, nullptr, kvidx, qbuf, kbuf, vbuf);

  // 3) proj^T f16
  proj_transpose_kernel<<<256, 256, 0, stream>>>(proj, projT);

  // 4) query = softmax(q @ proj), stored transposed (B-matrix for step 7)
  gemm_nt_wmma<MODE_F32><<<dim3(2, 2, NBATCH * NHEAD), 256, 0, stream>>>(
      qbuf, projT, 256, 256, 256, 65536, 0, 65536,
      f32scr, nullptr, nullptr, nullptr, nullptr, nullptr, nullptr);
  softmax256_kernel<<<NBATCH * NHEAD * 256, 256, 0, stream>>>(f32scr, queryT, 1);

  // 5) keym = softmax(k^T @ proj), row-major (A-matrix for step 6)
  gemm_nt_wmma<MODE_F32><<<dim3(2, 2, NBATCH * NHEAD), 256, 0, stream>>>(
      kbuf, projT, 256, 256, 256, 65536, 0, 65536,
      f32scr, nullptr, nullptr, nullptr, nullptr, nullptr, nullptr);
  softmax256_kernel<<<NBATCH * NHEAD * 256, 256, 0, stream>>>(f32scr, keym, 0);

  // 6) scores = keym @ v   (B = v^T, f16 output)
  gemm_nt_wmma<MODE_F16><<<dim3(2, 2, NBATCH * NHEAD), 256, 0, stream>>>(
      keym, vbuf, 256, 256, 256, 65536, 65536, 65536,
      nullptr, scores16, nullptr, nullptr, nullptr, nullptr, nullptr);

  // 7) attn = scores @ query (B = query^T), write [b, token, h, d] f16
  gemm_nt_wmma<MODE_ATTN><<<dim3(2, 2, NBATCH * NHEAD), 256, 0, stream>>>(
      scores16, queryT, 256, 256, 256, 65536, 65536, 0,
      nullptr, attnA, nullptr, nullptr, nullptr, nullptr, nullptr);

  // 8) O-proj + residual:  out = hidden + attn @ o_w^T
  f32_to_f16_kernel<<<4096, 256, 0, stream>>>(o_w, W16, (long)DIMX * DIMX);
  gemm_nt_wmma<MODE_RESID><<<dim3(DIMX / 128, ROWS / 128, 1), 256, 0, stream>>>(
      attnA, W16, ROWS, DIMX, DIMX, 0, 0, 0,
      out, nullptr, hidden, nullptr, nullptr, nullptr, nullptr);

  // 9) rmsnorm2 -> FFF tree, out += y
  rmsnorm_kernel<float><<<ROWS, 256, 0, stream>>>(out, ln2, f32scr);
  fff_kernel<<<ROWS, 256, 0, stream>>>(f32scr, Xw, Yw, out);
}
